// BigBirdRegressor_42150809043587
// MI455X (gfx1250) — matrix operand order repaired and checked by reading, not compiled, not run on hardware
//
#include <hip/hip_runtime.h>

// ---------------------------------------------------------------------------
// BigBird forward for MI455X (gfx1250, wave32, WMMA).
// All GEMMs + attention einsums use v_wmma_f32_16x16x32_f16 (f16 in, f32 acc).
// Weights are pre-transposed to [Nout][K] so all WMMA B-fragments are
// contiguous in LDS (ds_load_b128). Tile staging uses the gfx1250 async
// global->LDS path (global_load_async_to_lds_b128 + s_wait_asynccnt).
// ---------------------------------------------------------------------------

typedef _Float16 half_t;
typedef __attribute__((ext_vector_type(16))) _Float16 v16h;
typedef __attribute__((ext_vector_type(8)))  float    v8f;
typedef __attribute__((ext_vector_type(4)))  int      v4i;

constexpr int BB     = 2;
constexpr int SEQ    = 4096;
constexpr int DIM    = 768;
constexpr int NH     = 12;
constexpr int DH     = 64;
constexpr int BS     = 64;
constexpr int NBLK   = SEQ / BS;   // 64
constexpr int RR     = 3;
constexpr int FFD    = 4 * DIM;    // 3072
constexpr int NLAY   = 2;
constexpr int MROWS  = BB * SEQ;   // 8192

// padded f16 row stride: 72 halves = 144 B (16B-aligned, 36-bank stride)
constexpr int PAD16  = 72;

#ifndef __has_builtin
#define __has_builtin(x) 0
#endif
#if __has_builtin(__builtin_amdgcn_global_load_async_to_lds_b128) && \
    __has_builtin(__builtin_amdgcn_s_wait_asynccnt)
#define USE_ASYNC_COPY 1
#else
#define USE_ASYNC_COPY 0
#endif

#if USE_ASYNC_COPY
// param types per compiler diagnostic: v4i AS(1)* (printed __device__*) and
// the LDS-side counterpart in AS(3)
typedef __attribute__((address_space(1))) v4i gv4i;
typedef __attribute__((address_space(3))) v4i lv4i;
#endif

// copy 16 bytes global -> LDS (async engine when available)
static __device__ __forceinline__ void cp16(const half_t* g, half_t* l) {
#if USE_ASYNC_COPY
  __builtin_amdgcn_global_load_async_to_lds_b128((gv4i*)(void*)g,
                                                 (lv4i*)(void*)l, 0, 0);
#else
  *(float4*)l = *(const float4*)g;
#endif
}
static __device__ __forceinline__ void cp16_wait() {
#if USE_ASYNC_COPY
  __builtin_amdgcn_s_wait_asynccnt(0);
#endif
}

static __device__ __forceinline__ v8f wmma32(v16h a, v16h b, v8f c) {
  return __builtin_amdgcn_wmma_f32_16x16x32_f16(false, a, false, b, (short)0, c,
                                                false, false);
}

static __device__ __forceinline__ float gelu_new(float x) {
  const float c = 0.7978845608028654f;
  return 0.5f * x * (1.0f + tanhf(c * (x + 0.044715f * x * x * x)));
}

// ---------------------------------------------------------------------------
// f32 -> f16 conversion with transpose: src[K][Nout] -> dst[Nout][K]
// ---------------------------------------------------------------------------
__global__ __launch_bounds__(256) void f32_to_f16_t_kernel(
    const float* __restrict__ src, half_t* __restrict__ dst, int K, int Nout) {
  __shared__ float tile[32][33];
  int bx = blockIdx.x * 32;           // Nout base
  int by = blockIdx.y * 32;           // K base
  int tx = threadIdx.x & 31, ty = threadIdx.x >> 5;  // 32 x 8
#pragma unroll
  for (int j = 0; j < 32; j += 8)
    tile[ty + j][tx] = src[(size_t)(by + ty + j) * Nout + (bx + tx)];
  __syncthreads();
#pragma unroll
  for (int j = 0; j < 32; j += 8)
    dst[(size_t)(bx + ty + j) * K + (by + tx)] = (half_t)tile[tx][ty + j];
}

// ---------------------------------------------------------------------------
// Tiled WMMA GEMM: out[M x Nout] = A[M x K] (f16) * BwT[Nout x K] (f16, W^T)
// 64x64 block, BK=32, 128 threads (4 waves), each wave does 16 rows x 64 cols.
// epi: 0 = plain f32 out; 1 = +bias, f16 out; 2 = +bias+gelu, f16 out
// ---------------------------------------------------------------------------
__global__ __launch_bounds__(128) void gemm_wmma_kernel(
    const half_t* __restrict__ A, const half_t* __restrict__ BwT,
    const float* __restrict__ bias,
    float* __restrict__ outf, half_t* __restrict__ outh,
    int Mrows, int K, int Nout, int epi) {
  __shared__ half_t sA[64][PAD16];    // rows of A   (only [0,32) used per step)
  __shared__ half_t sBT[64][PAD16];   // cols of W   (only [0,32) used per step)

  const int tid    = threadIdx.x;
  const int lane   = tid & 31;
  const int wave   = tid >> 5;           // 0..3
  const int half16 = (lane >= 16) ? 1 : 0;
  const int l15    = lane & 15;
  const int bm     = blockIdx.y * 64;
  const int bn     = blockIdx.x * 64;

  // per-thread staging slots: rows ar and ar+32, 8-half column chunk ac
  const int ar = tid >> 2;
  const int ac = (tid & 3) << 3;
  const half_t* ag0 = A   + (size_t)(bm + ar) * K + ac;
  const half_t* ag1 = A   + (size_t)(bm + ar + 32) * K + ac;
  const half_t* bg0 = BwT + (size_t)(bn + ar) * K + ac;
  const half_t* bg1 = BwT + (size_t)(bn + ar + 32) * K + ac;
  half_t* al0 = &sA[ar][ac];
  half_t* al1 = &sA[ar + 32][ac];
  half_t* bl0 = &sBT[ar][ac];
  half_t* bl1 = &sBT[ar + 32][ac];

  v8f zero8 = {0, 0, 0, 0, 0, 0, 0, 0};
  v8f acc[4];
  acc[0] = zero8; acc[1] = zero8; acc[2] = zero8; acc[3] = zero8;

  for (int k0 = 0; k0 < K; k0 += 32) {
    // stage A/B tiles: all loads issued before stores (or async engine)
    cp16(ag0, al0);
    cp16(ag1, al1);
    cp16(bg0, bl0);
    cp16(bg1, bl1);
    ag0 += 32; ag1 += 32; bg0 += 32; bg1 += 32;

    // prefetch next K-step tiles into cache (global_prefetch_b8)
    if (k0 + 32 < K) {
      if (tid < 64)
        __builtin_prefetch((const void*)(A + (size_t)(bm + tid) * K + k0 + 32),
                           0, 0);
      else
        __builtin_prefetch(
            (const void*)(BwT + (size_t)(bn + (tid - 64)) * K + k0 + 32), 0, 0);
    }
    cp16_wait();
    __syncthreads();

    // A fragment: halves 0-7 -> K=ka..ka+7, halves 8-15 -> K=ka+16..ka+23
    v16h af;
    {
      int mrow = wave * 16 + l15;
      int ka   = half16 * 8;
#pragma unroll
      for (int h2 = 0; h2 < 16; ++h2) {
        int kk = ((h2 < 8) ? h2 : h2 + 8) + ka;
        af[h2] = sA[mrow][kk];
      }
    }
#pragma unroll
    for (int t = 0; t < 4; ++t) {
      v16h bf;
      int col = t * 16 + l15;
      int kb  = half16 * 8;
#pragma unroll
      for (int h2 = 0; h2 < 16; ++h2) bf[h2] = sBT[col][kb + ((h2 < 8) ? h2 : h2 + 8)];  // table order
      acc[t] = wmma32(af, bf, acc[t]);
    }
    __syncthreads();
  }

  // epilogue
#pragma unroll
  for (int t = 0; t < 4; ++t) {
    int col  = bn + t * 16 + l15;
    float bv = (epi != 0) ? bias[col] : 0.0f;
#pragma unroll
    for (int r = 0; r < 8; ++r) {
      int row  = bm + wave * 16 + r + half16 * 8;
      float vv = acc[t][r] + bv;
      if (epi == 2) vv = gelu_new(vv);
      if (epi == 0) outf[(size_t)row * Nout + col] = vv;
      else          outh[(size_t)row * Nout + col] = (half_t)vv;
    }
  }
}

// ---------------------------------------------------------------------------
// Block-sparse flash attention. One workgroup (256 thr = 8 waves) per
// (batch, head, query-block). Middle blocks: 8 gathered key blocks
// {0, NB-1, m-1, m, m+1, rand0..2} (duplicates kept, matching reference).
// Edge blocks (m==0, m==NB-1): all 64 key blocks (full attention).
// Online softmax: running m/l in LDS, O accumulators in VGPRs.
// V staged transposed (sVT[dh][key]) so PV B-fragments are contiguous.
// ---------------------------------------------------------------------------
static __device__ __forceinline__ int key_block_of(int kb, bool edge, int mqb,
                                                   const int* rb, int h) {
  if (edge) return kb;
  switch (kb) {
    case 0: return 0;
    case 1: return NBLK - 1;
    case 2: return mqb - 1;
    case 3: return mqb;
    case 4: return mqb + 1;
    default: return rb[((size_t)h * NBLK + mqb) * RR + (kb - 5)];
  }
}

__global__ __launch_bounds__(256) void attn_kernel(
    const half_t* __restrict__ q, const half_t* __restrict__ k,
    const half_t* __restrict__ v, const int* __restrict__ rb,
    half_t* __restrict__ ctx) {
  __shared__ half_t sQ[64][PAD16];    // [qrow][dh]
  __shared__ half_t sK[64][PAD16];    // [key][dh]
  __shared__ half_t sVT[64][PAD16];   // [dh][key]  (transposed V)
  __shared__ float  sS[64][65];       // padded: row scan is conflict-free
  __shared__ half_t sP[64][PAD16];    // [qrow][key]
  __shared__ float  sred[4][64];      // softmax partial max/sum
  __shared__ float  sm_[64];
  __shared__ float  sl_[64];
  __shared__ float  sal[64];

  const int mqb  = blockIdx.x;
  const int h    = blockIdx.y;
  const int b    = blockIdx.z;
  const int tid  = threadIdx.x;
  const int lane = tid & 31;
  const int wave = tid >> 5;             // 0..7
  const int half16 = (lane >= 16) ? 1 : 0;
  const int l15  = lane & 15;

  // staging slots: rows sr, sr+32; 8-half column chunk sc
  const int sr = tid >> 3;
  const int sc = (tid & 7) << 3;

  const half_t* qbase = q + ((size_t)(b * SEQ + mqb * BS)) * DIM + h * DH;
  cp16(qbase + (size_t)sr * DIM + sc,        &sQ[sr][sc]);
  cp16(qbase + (size_t)(sr + 32) * DIM + sc, &sQ[sr + 32][sc]);
  if (tid < 64) { sm_[tid] = -1e30f; sl_[tid] = 0.0f; }

  v8f zero8 = {0, 0, 0, 0, 0, 0, 0, 0};
  v8f oacc[2];
  oacc[0] = zero8; oacc[1] = zero8;

  const bool edge = (mqb == 0) || (mqb == NBLK - 1);
  const int  nkb  = edge ? NBLK : 8;
  const int  r64  = tid & 63;            // softmax row
  const int  qd   = tid >> 6;            // softmax column quad (0..3)

  cp16_wait();
  __syncthreads();

  for (int kb = 0; kb < nkb; ++kb) {
    const int kidx = key_block_of(kb, edge, mqb, rb, h);

    const half_t* kbp = k + ((size_t)(b * SEQ + kidx * BS)) * DIM + h * DH;
    const half_t* vbp = v + ((size_t)(b * SEQ + kidx * BS)) * DIM + h * DH;

    // V loads issued first (manual; must transpose), K via async path
    float4 v0 = *(const float4*)(vbp + (size_t)sr * DIM + sc);
    float4 v1 = *(const float4*)(vbp + (size_t)(sr + 32) * DIM + sc);
    cp16(kbp + (size_t)sr * DIM + sc,        &sK[sr][sc]);
    cp16(kbp + (size_t)(sr + 32) * DIM + sc, &sK[sr + 32][sc]);
    {
      const half_t* h0 = (const half_t*)&v0;
      const half_t* h1 = (const half_t*)&v1;
#pragma unroll
      for (int j = 0; j < 8; ++j) {
        sVT[sc + j][sr]      = h0[j];
        sVT[sc + j][sr + 32] = h1[j];
      }
    }
    // prefetch next key/value block (global_prefetch_b8)
    if (kb + 1 < nkb) {
      int nidx = key_block_of(kb + 1, edge, mqb, rb, h);
      size_t nbase = ((size_t)(b * SEQ + nidx * BS)) * DIM + h * DH;
      if (tid < 64)
        __builtin_prefetch((const void*)(k + nbase + (size_t)r64 * DIM), 0, 0);
      else if (tid < 128)
        __builtin_prefetch((const void*)(v + nbase + (size_t)r64 * DIM), 0, 0);
    }
    cp16_wait();
    __syncthreads();

    // S = scale * Q K^T  (16 16x16 tiles; wave owns 2)
#pragma unroll
    for (int i2 = 0; i2 < 2; ++i2) {
      int tt = wave * 2 + i2;
      int mt = tt >> 2, kt = tt & 3;
      v8f sacc = zero8;
#pragma unroll
      for (int kk = 0; kk < 2; ++kk) {
        v16h af, bf;
        int mrow = mt * 16 + l15;
        int ka   = half16 * 8 + kk * 32;
#pragma unroll
        for (int h2 = 0; h2 < 16; ++h2)
          af[h2] = sQ[mrow][((h2 < 8) ? h2 : h2 + 8) + ka];
        int key = kt * 16 + l15;
        int kbb = half16 * 8 + kk * 32;
#pragma unroll
        for (int h2 = 0; h2 < 16; ++h2)
          bf[h2] = sK[key][((h2 < 8) ? h2 : h2 + 8) + kbb];        // table order
        sacc = wmma32(af, bf, sacc);
      }
#pragma unroll
      for (int r = 0; r < 8; ++r)
        sS[mt * 16 + r + half16 * 8][kt * 16 + l15] = sacc[r] * 0.125f;
    }
    __syncthreads();

    // online softmax: 4 threads per row, 16 columns each
    {
      float pm = -1e30f;
#pragma unroll
      for (int j = 0; j < 16; ++j) pm = fmaxf(pm, sS[r64][qd * 16 + j]);
      sred[qd][r64] = pm;
    }
    __syncthreads();
    float mnew = fmaxf(fmaxf(fmaxf(sred[0][r64], sred[1][r64]),
                             fmaxf(sred[2][r64], sred[3][r64])),
                       sm_[r64]);
    float ps = 0.0f;
#pragma unroll
    for (int j = 0; j < 16; ++j) {
      float p = __expf(sS[r64][qd * 16 + j] - mnew);
      ps += p;
      sP[r64][qd * 16 + j] = (half_t)p;
    }
    __syncthreads();
    sred[qd][r64] = ps;
    __syncthreads();
    if (qd == 0) {
      float ssum  = sred[0][r64] + sred[1][r64] + sred[2][r64] + sred[3][r64];
      float alpha = __expf(sm_[r64] - mnew);
      sl_[r64] = sl_[r64] * alpha + ssum;
      sm_[r64] = mnew;
      sal[r64] = alpha;
    }
    __syncthreads();

    // O = O*alpha + P V
#pragma unroll
    for (int i2 = 0; i2 < 2; ++i2) {
      int tt = wave * 2 + i2;
      int mt = tt >> 2, nt = tt & 3;
      int rowoff = mt * 16 + half16 * 8;
#pragma unroll
      for (int r = 0; r < 8; ++r) oacc[i2][r] *= sal[rowoff + r];
#pragma unroll
      for (int kk = 0; kk < 2; ++kk) {
        v16h af, bf;
        int mrow = mt * 16 + l15;
        int ka   = half16 * 8 + kk * 32;
#pragma unroll
        for (int h2 = 0; h2 < 16; ++h2)
          af[h2] = sP[mrow][((h2 < 8) ? h2 : h2 + 8) + ka];
        int dcol = nt * 16 + l15;
        int kbb  = half16 * 8 + kk * 32;
#pragma unroll
        for (int h2 = 0; h2 < 16; ++h2)
          bf[h2] = sVT[dcol][((h2 < 8) ? h2 : h2 + 8) + kbb];      // table order (V^T)
        oacc[i2] = wmma32(af, bf, oacc[i2]);
      }
    }
    __syncthreads();
  }

  // normalize and write ctx (f16, [B*N, D] with head column slice)
#pragma unroll
  for (int i2 = 0; i2 < 2; ++i2) {
    int tt = wave * 2 + i2;
    int mt = tt >> 2, nt = tt & 3;
#pragma unroll
    for (int r = 0; r < 8; ++r) {
      int row  = mt * 16 + r + half16 * 8;
      float ov = oacc[i2][r] / sl_[row];
      ctx[((size_t)(b * SEQ + mqb * BS + row)) * DIM + h * DH + nt * 16 + l15] =
          (half_t)ov;
    }
  }
}

// ---------------------------------------------------------------------------
// LayerNorm kernels (one workgroup per row; 768 = 3 * 256)
// ---------------------------------------------------------------------------
__global__ __launch_bounds__(256) void embed_ln_kernel(
    const float* __restrict__ inp, const float* __restrict__ pos,
    const float* __restrict__ tte, const float* __restrict__ g,
    const float* __restrict__ bta, float* __restrict__ Xo,
    half_t* __restrict__ Xh) {
  int row = blockIdx.x;
  int tid = threadIdx.x;
  const float* xin = inp + (size_t)row * DIM;
  const float* pe  = pos + (size_t)(row % SEQ) * DIM;
  float loc[3];
  float s = 0.f;
#pragma unroll
  for (int i = 0; i < 3; ++i) {
    int d = tid + i * 256;
    float vv = xin[d] + pe[d] + tte[d];
    loc[i] = vv; s += vv;
  }
  __shared__ float red[256];
  red[tid] = s; __syncthreads();
  for (int o = 128; o > 0; o >>= 1) {
    if (tid < o) red[tid] += red[tid + o];
    __syncthreads();
  }
  float mu = red[0] * (1.0f / DIM);
  __syncthreads();
  float s2 = 0.f;
#pragma unroll
  for (int i = 0; i < 3; ++i) { float dm = loc[i] - mu; s2 += dm * dm; }
  red[tid] = s2; __syncthreads();
  for (int o = 128; o > 0; o >>= 1) {
    if (tid < o) red[tid] += red[tid + o];
    __syncthreads();
  }
  float rstd = rsqrtf(red[0] * (1.0f / DIM) + 1e-12f);
#pragma unroll
  for (int i = 0; i < 3; ++i) {
    int d = tid + i * 256;
    float y = (loc[i] - mu) * rstd * g[d] + bta[d];
    Xo[(size_t)row * DIM + d] = y;
    Xh[(size_t)row * DIM + d] = (half_t)y;
  }
}

__global__ __launch_bounds__(256) void residual_ln_kernel(
    const float* __restrict__ xres, const float* __restrict__ tmp,
    const float* __restrict__ bias, const float* __restrict__ g,
    const float* __restrict__ bta, float* __restrict__ Xo,
    half_t* __restrict__ Xh) {
  int row = blockIdx.x;
  int tid = threadIdx.x;
  float loc[3];
  float s = 0.f;
#pragma unroll
  for (int i = 0; i < 3; ++i) {
    int d = tid + i * 256;
    float vv = xres[(size_t)row * DIM + d] + tmp[(size_t)row * DIM + d] + bias[d];
    loc[i] = vv; s += vv;
  }
  __shared__ float red[256];
  red[tid] = s; __syncthreads();
  for (int o = 128; o > 0; o >>= 1) {
    if (tid < o) red[tid] += red[tid + o];
    __syncthreads();
  }
  float mu = red[0] * (1.0f / DIM);
  __syncthreads();
  float s2 = 0.f;
#pragma unroll
  for (int i = 0; i < 3; ++i) { float dm = loc[i] - mu; s2 += dm * dm; }
  red[tid] = s2; __syncthreads();
  for (int o = 128; o > 0; o >>= 1) {
    if (tid < o) red[tid] += red[tid + o];
    __syncthreads();
  }
  float rstd = rsqrtf(red[0] * (1.0f / DIM) + 1e-12f);
#pragma unroll
  for (int i = 0; i < 3; ++i) {
    int d = tid + i * 256;
    float y = (loc[i] - mu) * rstd * g[d] + bta[d];
    Xo[(size_t)row * DIM + d] = y;
    Xh[(size_t)row * DIM + d] = (half_t)y;
  }
}

// ---------------------------------------------------------------------------
// Mean pool over sequence, final linear head
// ---------------------------------------------------------------------------
__global__ __launch_bounds__(256) void pool_kernel(const float* __restrict__ X,
                                                   float* __restrict__ pooled) {
  int idx = blockIdx.x;  // b*DIM + d
  int tid = threadIdx.x;
  int b = idx / DIM, d = idx % DIM;
  float s = 0.f;
  for (int n = tid; n < SEQ; n += 256)
    s += X[((size_t)(b * SEQ + n)) * DIM + d];
  __shared__ float red[256];
  red[tid] = s; __syncthreads();
  for (int o = 128; o > 0; o >>= 1) {
    if (tid < o) red[tid] += red[tid + o];
    __syncthreads();
  }
  if (tid == 0) pooled[idx] = red[0] * (1.0f / SEQ);
}

__global__ __launch_bounds__(256) void final_kernel(
    const float* __restrict__ pooled, const float* __restrict__ fw,
    const float* __restrict__ fb, float* __restrict__ out) {
  int tid = threadIdx.x;
  __shared__ float red[256];
  for (int b = 0; b < BB; ++b) {
    float s = 0.f;
    for (int d = tid; d < DIM; d += 256) s += pooled[b * DIM + d] * fw[d];
    red[tid] = s; __syncthreads();
    for (int o = 128; o > 0; o >>= 1) {
      if (tid < o) red[tid] += red[tid + o];
      __syncthreads();
    }
    if (tid == 0) out[b] = red[0] + fb[0];
    __syncthreads();
  }
}

// ---------------------------------------------------------------------------
// Host launcher
// ---------------------------------------------------------------------------
extern "C" void kernel_launch(void* const* d_in, const int* in_sizes, int n_in,
                              void* d_out, int out_size, void* d_ws,
                              size_t ws_size, hipStream_t stream) {
  (void)in_sizes; (void)n_in; (void)out_size; (void)ws_size;
  const float* inputs_embeds = (const float*)d_in[0];
  const int*   rand_blocks   = (const int*)d_in[1];
  const float* pos_emb = (const float*)d_in[2];
  const float* tt_emb  = (const float*)d_in[3];
  const float* emb_g   = (const float*)d_in[4];
  const float* emb_b   = (const float*)d_in[5];
  const float* Wq = (const float*)d_in[6];
  const float* bq = (const float*)d_in[7];
  const float* Wk = (const float*)d_in[8];
  const float* bk = (const float*)d_in[9];
  const float* Wv = (const float*)d_in[10];
  const float* bv = (const float*)d_in[11];
  const float* Wo = (const float*)d_in[12];
  const float* bo = (const float*)d_in[13];
  const float* ln1_g = (const float*)d_in[14];
  const float* ln1_b = (const float*)d_in[15];
  const float* Wi = (const float*)d_in[16];
  const float* bi = (const float*)d_in[17];
  const float* Wd = (const float*)d_in[18];
  const float* bd = (const float*)d_in[19];
  const float* ln2_g = (const float*)d_in[20];
  const float* ln2_b = (const float*)d_in[21];
  const float* fc_w = (const float*)d_in[22];
  const float* fc_b = (const float*)d_in[23];
  float* out = (float*)d_out;

  char* ws = (char*)d_ws;
  size_t off = 0;
  auto carve = [&](size_t bytes) -> void* {
    void* p = (void*)(ws + off);
    off += (bytes + 255) & ~(size_t)255;
    return p;
  };
  constexpr size_t DD   = (size_t)DIM * DIM;
  constexpr size_t DF   = (size_t)DIM * FFD;
  constexpr size_t ACT  = (size_t)MROWS * DIM;
  constexpr size_t ACTF = (size_t)MROWS * FFD;
  constexpr size_t LAYER_W = 4 * DD + 2 * DF;

  float*  X    = (float*)carve(ACT * 4);
  float*  Aact = (float*)carve(ACT * 4);
  float*  TMP  = (float*)carve(ACT * 4);
  half_t* ACTh = (half_t*)carve(ACT * 2);
  half_t* QH   = (half_t*)carve(ACT * 2);
  half_t* KH   = (half_t*)carve(ACT * 2);
  half_t* VH   = (half_t*)carve(ACT * 2);
  half_t* CTXh = (half_t*)carve(ACT * 2);
  half_t* HH   = (half_t*)carve(ACTF * 2);
  half_t* W16  = (half_t*)carve(NLAY * LAYER_W * 2);
  float*  POOL = (float*)carve((size_t)BB * DIM * 4);

  // transposed f16 weights: dst layout [Nout][K]
  auto cvtT = [&](const float* src, half_t* dst, int K, int Nout) {
    f32_to_f16_t_kernel<<<dim3(Nout / 32, K / 32), dim3(256), 0, stream>>>(
        src, dst, K, Nout);
  };
  for (int l = 0; l < NLAY; ++l) {
    half_t* base = W16 + (size_t)l * LAYER_W;
    cvtT(Wq + (size_t)l * DD, base,               DIM, DIM);
    cvtT(Wk + (size_t)l * DD, base + DD,          DIM, DIM);
    cvtT(Wv + (size_t)l * DD, base + 2 * DD,      DIM, DIM);
    cvtT(Wo + (size_t)l * DD, base + 3 * DD,      DIM, DIM);
    cvtT(Wi + (size_t)l * DF, base + 4 * DD,      DIM, FFD);
    cvtT(Wd + (size_t)l * DF, base + 4 * DD + DF, FFD, DIM);
  }

  embed_ln_kernel<<<dim3(MROWS), dim3(256), 0, stream>>>(
      inputs_embeds, pos_emb, tt_emb, emb_g, emb_b, X, ACTh);

  auto gemm = [&](const half_t* Ain, const half_t* BwT, const float* bias,
                  float* of, half_t* oh, int Mr, int K, int Nout, int epi) {
    gemm_wmma_kernel<<<dim3(Nout / 64, Mr / 64), dim3(128), 0, stream>>>(
        Ain, BwT, bias, of, oh, Mr, K, Nout, epi);
  };

  for (int l = 0; l < NLAY; ++l) {
    half_t* base = W16 + (size_t)l * LAYER_W;
    const half_t* wq16 = base;
    const half_t* wk16 = base + DD;
    const half_t* wv16 = base + 2 * DD;
    const half_t* wo16 = base + 3 * DD;
    const half_t* wi16 = base + 4 * DD;
    const half_t* wd16 = base + 4 * DD + DF;

    // Q/K/V projections (bias fused, f16 out)
    gemm(ACTh, wq16, bq + (size_t)l * DIM, nullptr, QH, MROWS, DIM, DIM, 1);
    gemm(ACTh, wk16, bk + (size_t)l * DIM, nullptr, KH, MROWS, DIM, DIM, 1);
    gemm(ACTh, wv16, bv + (size_t)l * DIM, nullptr, VH, MROWS, DIM, DIM, 1);

    // block-sparse attention
    attn_kernel<<<dim3(NBLK, NH, BB), dim3(256), 0, stream>>>(
        QH, KH, VH, rand_blocks + (size_t)l * NH * NBLK * RR, CTXh);

    // output projection (bo folded into LN), residual + LN1
    gemm(CTXh, wo16, nullptr, TMP, nullptr, MROWS, DIM, DIM, 0);
    residual_ln_kernel<<<dim3(MROWS), dim3(256), 0, stream>>>(
        X, TMP, bo + (size_t)l * DIM, ln1_g + (size_t)l * DIM,
        ln1_b + (size_t)l * DIM, Aact, ACTh);

    // FFN: gelu(a@Wi+bi) then @Wd (bd folded into LN), residual + LN2
    gemm(ACTh, wi16, bi + (size_t)l * FFD, nullptr, HH, MROWS, DIM, FFD, 2);
    gemm(HH, wd16, nullptr, TMP, nullptr, MROWS, FFD, DIM, 0);
    residual_ln_kernel<<<dim3(MROWS), dim3(256), 0, stream>>>(
        Aact, TMP, bd + (size_t)l * DIM, ln2_g + (size_t)l * DIM,
        ln2_b + (size_t)l * DIM, X, ACTh);
  }

  pool_kernel<<<dim3(BB * DIM), dim3(256), 0, stream>>>(X, POOL);
  final_kernel<<<dim3(1), dim3(256), 0, stream>>>(POOL, fc_w, fc_b, out);
}
